// AoEMoE_72438918414736
// MI455X (gfx1250) — matrix operand order repaired and checked by reading, not compiled, not run on hardware
//
#include <hip/hip_runtime.h>

// ---------------------------------------------------------------------------
// Types
// ---------------------------------------------------------------------------
typedef __bf16 bf16_t;
typedef __attribute__((ext_vector_type(16))) __bf16 v16bf;
typedef __attribute__((ext_vector_type(8)))  __bf16 v8bf;   // 16 bytes
typedef __attribute__((ext_vector_type(4)))  __bf16 v4bf;   // 8 bytes
typedef __attribute__((ext_vector_type(8)))  float  v8f;
typedef __attribute__((ext_vector_type(4)))  float  v4f;
typedef __attribute__((ext_vector_type(4)))  unsigned int v4u;
typedef __attribute__((ext_vector_type(8)))  int    v8i;
typedef __attribute__((ext_vector_type(4)))  int    v4i;

#define BM 128            // block M (tokens)
#define BN 256            // block N (out features)
#define BK 32             // K step = one WMMA K
#define THREADS 256       // 8 waves: 2 (M) x 4 (N), each wave 64x64 output

// LDS element offsets inside the single 48KB shared block:
//   A0 @ 0      (128x32 =  8KB)   A1 @ 4096  elems
//   W0 @ 8192   (256x32 = 16KB)   W1 @ 16384 elems
#define A_ELEMS 4096
#define W_ELEMS 8192
#define LDS_ELEMS (2 * A_ELEMS + 2 * W_ELEMS)   // 24576 elems = 48KB

// Problem sizes (fixed by the reference)
#define NE 8
#define HH 1024
#define FF 2048
#define RR 64
#define NN 4096      // B*S
#define ER 512       // E*R

#if __has_builtin(__builtin_amdgcn_tensor_load_to_lds)
#define HAVE_TDM 1
#else
#define HAVE_TDM 0
#endif

#if __has_builtin(__builtin_amdgcn_s_wait_tensorcnt)
#define WAIT_TENSORCNT(n) __builtin_amdgcn_s_wait_tensorcnt(n)
#else
#define WAIT_TENSORCNT(n) asm volatile("s_wait_tensorcnt " #n ::: "memory")
#endif

// ---------------------------------------------------------------------------
// The TDM builtin takes the LDS destination as an *integer* offset, so the
// __shared__ array has no IR-visible writers and LLVM folds its loads to
// undef.  Escaping the pointer into an asm with a memory clobber makes the
// compiler treat the array as externally written (zero instructions emitted).
// ---------------------------------------------------------------------------
__device__ __forceinline__ void lds_publish(bf16_t* p)
{
    asm volatile("" : : "v"(p) : "memory");
}

// ---------------------------------------------------------------------------
// TDM: DMA one [rows x 32] bf16 tile (row stride = ld elements) from global
// into LDS at byte offset lds_off (packed row-major, stride 32 elems).
// D# layout per CDNA5 ISA (2-D tile, groups 2/3 zero).
// ---------------------------------------------------------------------------
#if HAVE_TDM
__device__ __forceinline__ void tdm_tile(unsigned lds_off,
                                         const bf16_t* __restrict__ g,
                                         unsigned ld, unsigned rows)
{
    unsigned long long ga = (unsigned long long)(const void*)g;
    v4u g0;
    g0[0] = 1u;                                   // count=1, user-mode
    g0[1] = lds_off;                              // lds_addr (bytes)
    g0[2] = (unsigned)ga;                         // global_addr[31:0]
    g0[3] = (unsigned)(ga >> 32) | (2u << 30);    // global_addr[56:32] | type=2
    v8i g1;
    unsigned dim0 = ld;                           // tensor_dim0 = row length
    unsigned dim1 = 1u << 20;                     // tensor_dim1: ample for OOB
    g1[0] = (int)(1u << 16);                      // wg_mask=0, data_size=1 (2B)
    g1[1] = (int)((dim0 & 0xFFFFu) << 16);        // dim0[15:0] -> bits 63:48
    g1[2] = (int)((dim0 >> 16) | ((dim1 & 0xFFFFu) << 16));
    g1[3] = (int)((dim1 >> 16) | ((unsigned)BK << 16));   // tile_dim0 = 32
    g1[4] = (int)rows;                            // tile_dim1, tile_dim2 = 0
    g1[5] = (int)ld;                              // tensor_dim0_stride[31:0]
    g1[6] = 0;                                    // stride0[47:32] | stride1 lo
    g1[7] = 0;                                    // stride1 hi
    v4i z4 = {0, 0, 0, 0};
#if __clang_major__ >= 23
    v8i z8 = {0, 0, 0, 0, 0, 0, 0, 0};
    __builtin_amdgcn_tensor_load_to_lds(g0, g1, z4, z4, z8, 0);
#else
    __builtin_amdgcn_tensor_load_to_lds(g0, g1, z4, z4, 0);
#endif
}
#endif

// ---------------------------------------------------------------------------
// Fallback synchronous tile copy
// ---------------------------------------------------------------------------
template<int ROWS>
__device__ __forceinline__ void load_tile_sync(bf16_t* __restrict__ lds,
                                               const bf16_t* __restrict__ g,
                                               long ld, int tid)
{
#pragma unroll
    for (int i = 0; i < ROWS / 64; ++i) {
        int idx = tid + i * THREADS;
        int row = idx >> 2;
        int ch  = (idx & 3) * 8;
        *(v8bf*)(lds + row * BK + ch) = *(const v8bf*)(g + row * ld + ch);
    }
}

// ---------------------------------------------------------------------------
// WMMA fragments (wave32 layouts per CDNA5 ISA §7.12.2)
// ---------------------------------------------------------------------------
__device__ __forceinline__ v16bf frag_a(const bf16_t* t, int row, int half)
{
    const bf16_t* p = t + row * BK + half * 8;
    v8bf lo = *(const v8bf*)p;
    v8bf hi = *(const v8bf*)(p + 16);
    v16bf r;
#pragma unroll
    for (int i = 0; i < 8; ++i) { r[i] = lo[i]; r[i + 8] = hi[i]; }
    return r;
}

__device__ __forceinline__ v16bf frag_b(const bf16_t* t, int nrow, int half)
{
    const bf16_t* p = t + nrow * BK + half * 8;
    v8bf lo = *(const v8bf*)p;
    v8bf hi = *(const v8bf*)(p + 16);
    v16bf r;
#pragma unroll
    for (int i = 0; i < 8; ++i) { r[i] = lo[i]; r[i + 8] = hi[i]; }
    return r;
}

__device__ __forceinline__ v8f wmma_bf16(v16bf a, v16bf b, v8f c)
{
    return __builtin_amdgcn_wmma_f32_16x16x32_bf16(
        false, a, false, b, (short)0, c, false, false);
}

// One BK=32 K-step: wave computes a 64(M) x 64(N) tile = 4x4 WMMA tiles.
__device__ __forceinline__ void mma_step(const bf16_t* At, const bf16_t* Wt,
                                         int waveM, int waveN, int half, int r,
                                         v8f (&c)[4][4])
{
    v16bf a[4], b[4];
#pragma unroll
    for (int mt = 0; mt < 4; ++mt)
        a[mt] = frag_a(At, waveM * 64 + mt * 16 + r, half);
#pragma unroll
    for (int nt = 0; nt < 4; ++nt)
        b[nt] = frag_b(Wt, waveN * 64 + nt * 16 + r, half);
#pragma unroll
    for (int mt = 0; mt < 4; ++mt)
#pragma unroll
        for (int nt = 0; nt < 4; ++nt)
            c[mt][nt] = wmma_bf16(a[mt], b[nt], c[mt][nt]);
}

// ---------------------------------------------------------------------------
// Double-buffered K-loop: TDM fills one LDS tile pair while 8 waves run WMMA
// on the other.  LDS offsets are integer arithmetic on the single shared
// block so fragment reads stay ds_load (no flat fallback).
// ---------------------------------------------------------------------------
__device__ __forceinline__ void gemm_loop(
    bf16_t* __restrict__ lds,
    const bf16_t* __restrict__ A, long lda,
    const bf16_t* __restrict__ W, long ldw,
    int nk, int tid, int waveM, int waveN, int half, int r,
    v8f (&c)[4][4])
{
#if HAVE_TDM
    const bool issuer = (tid < 32);          // wave 0 drives the TDM
    if (issuer) {
        tdm_tile(0u,                 A, (unsigned)lda, BM);
        tdm_tile(2u * W_ELEMS,       W, (unsigned)ldw, BN);   // W0 @ 16384 B
    }
    for (int kt = 0; kt < nk; ++kt) {
        const int b = kt & 1;
        if (issuer) {
            if (kt + 1 < nk) {
                // Prefetch next pair into the alternate buffer, then wait
                // only for the current pair (<=2 leaves next pair in flight).
                const unsigned nb = (unsigned)(b ^ 1);
                tdm_tile(nb * (2u * A_ELEMS),
                         A + (long)(kt + 1) * BK, (unsigned)lda, BM);
                tdm_tile(2u * W_ELEMS + nb * (2u * W_ELEMS),
                         W + (long)(kt + 1) * BK, (unsigned)ldw, BN);
                WAIT_TENSORCNT(2);
            } else {
                WAIT_TENSORCNT(0);
            }
        }
        __syncthreads();
        lds_publish(lds);   // TDM wrote LDS: keep ds_loads alive & uncached
        mma_step(lds + b * A_ELEMS,
                 lds + 2 * A_ELEMS + b * W_ELEMS,
                 waveM, waveN, half, r, c);
        __syncthreads();
    }
#else
    for (int kt = 0; kt < nk; ++kt) {
        load_tile_sync<BM>(lds, A + (long)kt * BK, lda, tid);
        load_tile_sync<BN>(lds + 2 * A_ELEMS, W + (long)kt * BK, ldw, tid);
        __syncthreads();
        mma_step(lds, lds + 2 * A_ELEMS, waveM, waveN, half, r, c);
        __syncthreads();
    }
#endif
}

// ---------------------------------------------------------------------------
// Generic GEMM: C[f32] = A[bf16, M x K] * W[bf16, N x K]^T  (gate projection)
// ---------------------------------------------------------------------------
__global__ __launch_bounds__(THREADS)
void gemm_bf16_f32(const bf16_t* __restrict__ A, const bf16_t* __restrict__ W,
                   float* __restrict__ C, int K, long lda, long ldw, long ldc)
{
    __shared__ __align__(16) bf16_t lds[LDS_ELEMS];
    const int tid   = threadIdx.x;
    const int waveM = (tid >> 5) >> 2, waveN = (tid >> 5) & 3;
    const int half  = (tid >> 4) & 1,  r     = tid & 15;
    const long row0 = (long)blockIdx.y * BM;
    const long col0 = (long)blockIdx.x * BN;

    v8f c[4][4] = {};
    gemm_loop(lds, A + row0 * lda, lda, W + col0 * ldw, ldw,
              K / BK, tid, waveM, waveN, half, r, c);
#pragma unroll
    for (int mt = 0; mt < 4; ++mt)
#pragma unroll
        for (int nt = 0; nt < 4; ++nt)
#pragma unroll
            for (int v = 0; v < 8; ++v) {
                long row = row0 + waveM * 64 + mt * 16 + half * 8 + v;
                long col = col0 + waveN * 64 + nt * 16 + r;
                C[row * ldc + col] = c[mt][nt][v];
            }
}

// ---------------------------------------------------------------------------
// Routing: scores = ||gate_hidden||_2 per expert, top-2, softmax -> dense w.
// ---------------------------------------------------------------------------
__global__ void routing_kernel(const float* __restrict__ gh,
                               float* __restrict__ wfull)
{
    int n = blockIdx.x * blockDim.x + threadIdx.x;
    const float* g = gh + (long)n * ER;
    float s[NE];
#pragma unroll
    for (int e = 0; e < NE; ++e) {
        float acc = 0.f;
#pragma unroll
        for (int rr = 0; rr < RR; rr += 4) {
            v4f v = *(const v4f*)(g + e * RR + rr);
            acc += v.x * v.x + v.y * v.y + v.z * v.z + v.w * v.w;
        }
        s[e] = sqrtf(acc);
    }
    int i1 = 0; float v1 = s[0];
#pragma unroll
    for (int e = 1; e < NE; ++e)
        if (s[e] > v1) { v1 = s[e]; i1 = e; }
    int i2 = -1; float v2 = -3.4e38f;
#pragma unroll
    for (int e = 0; e < NE; ++e)
        if (e != i1 && s[e] > v2) { v2 = s[e]; i2 = e; }
    float w1 = 1.f / (1.f + __expf(v2 - v1));   // softmax over {v1, v2}
    float w2 = 1.f - w1;
#pragma unroll
    for (int e = 0; e < NE; ++e)
        wfull[(long)n * NE + e] = (e == i1) ? w1 : (e == i2) ? w2 : 0.f;
}

// ---------------------------------------------------------------------------
// Per-expert fused up & gate-lowrank GEMMs + silu epilogue:
//   h[n,f] = wfull[n,e] * silu(g[n,f]) * up[n,f]   (bf16 scratch)
// ---------------------------------------------------------------------------
__global__ __launch_bounds__(THREADS)
void moe_up_kernel(const bf16_t* __restrict__ xb,    // [4096,1024]
                   const bf16_t* __restrict__ ghb,   // [4096,512]
                   const bf16_t* __restrict__ wUb,   // [8,2048,1024]
                   const bf16_t* __restrict__ wBb,   // [8,2048,64]
                   const float*  __restrict__ wfull, // [4096,8]
                   bf16_t* __restrict__ h,           // [4096,2048]
                   int e)
{
    __shared__ __align__(16) bf16_t lds[LDS_ELEMS];
    const int tid   = threadIdx.x;
    const int waveM = (tid >> 5) >> 2, waveN = (tid >> 5) & 3;
    const int half  = (tid >> 4) & 1,  r     = tid & 15;
    const long row0 = (long)blockIdx.y * BM;   // token rows
    const long col0 = (long)blockIdx.x * BN;   // F cols
    const bf16_t* Wu = wUb + (long)e * FF * HH;
    const bf16_t* Wb = wBb + (long)e * FF * RR;

    // up = xb * W_up^T  (K = 1024)
    v8f cu[4][4] = {};
    gemm_loop(lds, xb + row0 * HH, HH, Wu + col0 * HH, HH,
              HH / BK, tid, waveM, waveN, half, r, cu);
    // g = gate_hidden[:,e,:] * W_B^T  (K = 64)
    v8f cg[4][4] = {};
    gemm_loop(lds, ghb + row0 * ER + (long)e * RR, ER, Wb + col0 * RR, RR,
              RR / BK, tid, waveM, waveN, half, r, cg);

#pragma unroll
    for (int mt = 0; mt < 4; ++mt)
#pragma unroll
        for (int nt = 0; nt < 4; ++nt)
#pragma unroll
            for (int v = 0; v < 8; ++v) {
                long row = row0 + waveM * 64 + mt * 16 + half * 8 + v;
                long col = col0 + waveN * 64 + nt * 16 + r;
                float wv = wfull[row * NE + e];
                float gv = cg[mt][nt][v];
                float uv = cu[mt][nt][v];
                float silu = gv / (1.f + __expf(-gv));
                h[row * FF + col] = (bf16_t)(wv * silu * uv);
            }
}

// ---------------------------------------------------------------------------
// Per-expert down projection, accumulated into out (experts serialized).
// ---------------------------------------------------------------------------
__global__ __launch_bounds__(THREADS)
void moe_down_kernel(const bf16_t* __restrict__ h,   // [4096,2048]
                     const bf16_t* __restrict__ wDb, // [8,1024,2048]
                     float* __restrict__ out,        // [4096,1024]
                     int e)
{
    __shared__ __align__(16) bf16_t lds[LDS_ELEMS];
    const int tid   = threadIdx.x;
    const int waveM = (tid >> 5) >> 2, waveN = (tid >> 5) & 3;
    const int half  = (tid >> 4) & 1,  r     = tid & 15;
    const long row0 = (long)blockIdx.y * BM;   // token rows
    const long col0 = (long)blockIdx.x * BN;   // H cols
    const bf16_t* Wd = wDb + (long)e * HH * FF;

    v8f c[4][4] = {};
    gemm_loop(lds, h + row0 * FF, FF, Wd + col0 * FF, FF,
              FF / BK, tid, waveM, waveN, half, r, c);
#pragma unroll
    for (int mt = 0; mt < 4; ++mt)
#pragma unroll
        for (int nt = 0; nt < 4; ++nt)
#pragma unroll
            for (int v = 0; v < 8; ++v) {
                long row = row0 + waveM * 64 + mt * 16 + half * 8 + v;
                long col = col0 + waveN * 64 + nt * 16 + r;
                out[row * HH + col] += c[mt][nt][v];
            }
}

// ---------------------------------------------------------------------------
// Elementwise helpers
// ---------------------------------------------------------------------------
__global__ void cvt_f32_bf16(const float* __restrict__ s,
                             bf16_t* __restrict__ d, long n)
{
    long i = ((long)blockIdx.x * blockDim.x + threadIdx.x) * 4;
    if (i >= n) return;
    v4f v = *(const v4f*)(s + i);
    v4bf o;
    o[0] = (bf16_t)v.x; o[1] = (bf16_t)v.y;
    o[2] = (bf16_t)v.z; o[3] = (bf16_t)v.w;
    *(v4bf*)(d + i) = o;
}

__global__ void zero_f32(float* __restrict__ p, long n)
{
    long i = ((long)blockIdx.x * blockDim.x + threadIdx.x) * 4;
    if (i >= n) return;
    v4f z = {};
    *(v4f*)(p + i) = z;
}

// ---------------------------------------------------------------------------
// Host launcher.  Workspace layout (~108 MB, 256B-aligned chunks).
// ---------------------------------------------------------------------------
extern "C" void kernel_launch(void* const* d_in, const int* in_sizes, int n_in,
                              void* d_out, int out_size, void* d_ws, size_t ws_size,
                              hipStream_t stream)
{
    (void)in_sizes; (void)n_in; (void)out_size; (void)ws_size;

    const float* x     = (const float*)d_in[0];  // [2,2048,1024]
    const float* W_A   = (const float*)d_in[1];  // [8,64,1024]
    const float* W_B   = (const float*)d_in[2];  // [8,2048,64]
    const float* W_up  = (const float*)d_in[3];  // [8,2048,1024]
    const float* W_dn  = (const float*)d_in[4];  // [8,1024,2048]
    float* out = (float*)d_out;                  // [2,2048,1024]

    char* w = (char*)d_ws;
    auto carve = [&](size_t bytes) {
        char* p = w;
        w += (bytes + 255) & ~(size_t)255;
        return p;
    };
    bf16_t* xb    = (bf16_t*)carve((size_t)NN * HH * 2);
    bf16_t* wAb   = (bf16_t*)carve((size_t)NE * RR * HH * 2);
    bf16_t* wBb   = (bf16_t*)carve((size_t)NE * FF * RR * 2);
    bf16_t* wUb   = (bf16_t*)carve((size_t)NE * FF * HH * 2);
    bf16_t* wDb   = (bf16_t*)carve((size_t)NE * HH * FF * 2);
    float*  gh    = (float*)carve((size_t)NN * ER * 4);
    bf16_t* ghb   = (bf16_t*)carve((size_t)NN * ER * 2);
    float*  wfull = (float*)carve((size_t)NN * NE * 4);
    bf16_t* hbuf  = (bf16_t*)carve((size_t)NN * FF * 2);

    auto cvt = [&](const float* s, bf16_t* d, long n) {
        cvt_f32_bf16<<<(unsigned)(n / 4 / 256), 256, 0, stream>>>(s, d, n);
    };
    cvt(x,    xb,  (long)NN * HH);
    cvt(W_A,  wAb, (long)NE * RR * HH);
    cvt(W_B,  wBb, (long)NE * FF * RR);
    cvt(W_up, wUb, (long)NE * FF * HH);
    cvt(W_dn, wDb, (long)NE * HH * FF);

    // gate_hidden = x * W_A^T  -> [4096, 512] f32
    {
        dim3 grid(ER / BN, NN / BM);   // 2 x 32
        gemm_bf16_f32<<<grid, THREADS, 0, stream>>>(xb, wAb, gh,
                                                    HH, HH, HH, ER);
    }
    cvt(gh, ghb, (long)NN * ER);
    routing_kernel<<<NN / 256, 256, 0, stream>>>(gh, wfull);

    zero_f32<<<(unsigned)((long)NN * HH / 4 / 256), 256, 0, stream>>>(
        out, (long)NN * HH);

    for (int e = 0; e < NE; ++e) {
        dim3 gu(FF / BN, NN / BM);     // 8 x 32
        moe_up_kernel<<<gu, THREADS, 0, stream>>>(xb, ghb, wUb, wBb,
                                                  wfull, hbuf, e);
        dim3 gd(HH / BN, NN / BM);     // 4 x 32
        moe_down_kernel<<<gd, THREADS, 0, stream>>>(hbuf, wDb, out, e);
    }
}
